// MolCLREncoder_84301618086280
// MI455X (gfx1250) — matrix operand order repaired and checked by reading, not compiled, not run on hardware
//
#include <hip/hip_runtime.h>

// ---------------------------------------------------------------------------
// Types for CDNA5 WMMA (wave32): v16bf A/B fragments, v8f accumulator.
// ---------------------------------------------------------------------------
typedef __attribute__((ext_vector_type(16))) __bf16 v16bf;
typedef __attribute__((ext_vector_type(8)))  float  v8f;

union FragBF {
    v16bf v;
    unsigned short u[16];
};

__device__ __forceinline__ unsigned short f32_to_bf16(float f) {
    unsigned int x = __float_as_uint(f);
    x += 0x7FFFu + ((x >> 16) & 1u);   // round-to-nearest-even
    return (unsigned short)(x >> 16);
}

// ---------------------------------------------------------------------------
// Elementwise / setup kernels
// ---------------------------------------------------------------------------
__global__ void zero_f32_kernel(float* __restrict__ p, long n) {
    long i = (long)blockIdx.x * blockDim.x + threadIdx.x;
    if (i < n) p[i] = 0.0f;
}

__global__ void deg_accum_kernel(const int* __restrict__ dst, float* __restrict__ deg, int E) {
    long e = (long)blockIdx.x * blockDim.x + threadIdx.x;
    if (e < E) atomicAdd(&deg[dst[e]], 1.0f);
}

__global__ void rsqrt_kernel(float* __restrict__ dis, int N) {
    long i = (long)blockIdx.x * blockDim.x + threadIdx.x;
    if (i < N) dis[i] = rsqrtf(dis[i] + 1.0f);   // deg includes self-loop
}

__global__ void norm_kernel(const int* __restrict__ src, const int* __restrict__ dst,
                            const float* __restrict__ dis, float* __restrict__ nrm, int E) {
    long e = (long)blockIdx.x * blockDim.x + threadIdx.x;
    if (e < E) nrm[e] = dis[src[e]] * dis[dst[e]];
}

// agg = hW * dis^2 + bias   (self-loop contribution + bias, pre-scatter)
__global__ void self_bias_kernel(const float* __restrict__ hW, const float* __restrict__ dis,
                                 const float* __restrict__ bias, float* __restrict__ agg,
                                 long total, int fshift, int fmask) {
    long idx = (long)blockIdx.x * blockDim.x + threadIdx.x;
    if (idx >= total) return;
    long i = idx >> fshift;
    int  f = (int)(idx & fmask);
    float d = dis[i];
    agg[idx] = hW[idx] * d * d + bias[f];
}

// agg[dst] += hW[src] * norm  over edges; one thread per (edge, float4 chunk)
__global__ void edge_scatter_kernel(const int* __restrict__ src, const int* __restrict__ dst,
                                    const float* __restrict__ nrm, const float* __restrict__ hW,
                                    float* __restrict__ agg, long total, int qshift, int qmask, int F) {
    long idx = (long)blockIdx.x * blockDim.x + threadIdx.x;
    if (idx >= total) return;
    int e  = (int)(idx >> qshift);
    int fq = (int)(idx & qmask);
    int s = src[e], d = dst[e];
    float nv = nrm[e];
    const float4 h4 = ((const float4*)(hW + (long)s * F))[fq];
    float* ap = agg + (long)d * F + (long)fq * 4;
    atomicAdd(ap + 0, h4.x * nv);
    atomicAdd(ap + 1, h4.y * nv);
    atomicAdd(ap + 2, h4.z * nv);
    atomicAdd(ap + 3, h4.w * nv);
}

__global__ void relu_kernel(float* __restrict__ p, long n) {
    long i = (long)blockIdx.x * blockDim.x + threadIdx.x;
    if (i < n) p[i] = fmaxf(p[i], 0.0f);
}

// pooled[batch[i]] += h[i]  (float4 chunks)
__global__ void pool_scatter_kernel(const float* __restrict__ h, const int* __restrict__ batch,
                                    float* __restrict__ pooled, long total, int qshift, int qmask, int F) {
    long idx = (long)blockIdx.x * blockDim.x + threadIdx.x;
    if (idx >= total) return;
    long i = idx >> qshift;
    int fq = (int)(idx & qmask);
    int g = batch[i];
    const float4 h4 = ((const float4*)(h + i * F))[fq];
    float* pp = pooled + (long)g * F + (long)fq * 4;
    atomicAdd(pp + 0, h4.x);
    atomicAdd(pp + 1, h4.y);
    atomicAdd(pp + 2, h4.z);
    atomicAdd(pp + 3, h4.w);
}

__global__ void count_kernel(const int* __restrict__ batch, float* __restrict__ cnt, int N) {
    long i = (long)blockIdx.x * blockDim.x + threadIdx.x;
    if (i < N) atomicAdd(&cnt[batch[i]], 1.0f);
}

__global__ void pool_div_kernel(float* __restrict__ pooled, const float* __restrict__ cnt,
                                long total, int fshift) {
    long idx = (long)blockIdx.x * blockDim.x + threadIdx.x;
    if (idx >= total) return;
    long g = idx >> fshift;
    pooled[idx] /= fmaxf(cnt[g], 1.0f);
}

// ---------------------------------------------------------------------------
// WMMA GEMM: C[M,F] = A[M,K] @ W[K,F] (+ bias), f32 in/out, bf16 tensor cores.
// Block = 256 threads = 8 waves. Tile: 128 (M) x 64 (N), K in steps of 32.
//
// B is staged in LDS *pre-swizzled into the WMMA fragment layout*
// (contiguous k order: (k,n) -> ctile=n>>4, lane=(n&15)+16*(k>>4), slot=k&15), so a
// fragment load is one 32-byte v16bf read per lane (ds_load_b128 x2) instead
// of 16 scalar ds_load_u16. Double-buffered: next W tile streams while the
// current WMMAs execute; one barrier per K-step.
//
// A fragments are loaded straight from global (each row feeds exactly one
// wave, LDS staging buys nothing): 4x float4 per lane + f32->bf16 in regs.
// ---------------------------------------------------------------------------
__global__ __launch_bounds__(256) void gemm_wmma_bf16_kernel(
    const float* __restrict__ A, const float* __restrict__ W,
    const float* __restrict__ bias, float* __restrict__ C,
    int M, int K, int F)
{
    // [buf][ctile][lane][slot] : 2 * 4 * 32 * 16 bf16 = 8 KB
    __shared__ __align__(32) unsigned short sB[2][4 * 32 * 16];

    const int tid   = threadIdx.x;
    const int wave  = tid >> 5;
    const int lane  = tid & 31;
    const int tileM = blockIdx.y * 128;
    const int tileN = blockIdx.x * 64;
    const bool kvec = ((K & 3) == 0);

    v8f acc[4];
#pragma unroll
    for (int c = 0; c < 4; ++c)
#pragma unroll
        for (int j = 0; j < 8; ++j) acc[c][j] = 0.0f;

    // Stage one 32xK-slice of W into fragment-layout LDS buffer `buf`.
    auto stageB = [&](int buf, int kt) {
        for (int p = tid; p < 1024; p += 256) {     // 16 k-pairs x 64 cols
            int kp = p >> 6;                        // k-pair index 0..15
            int n  = p & 63;
            int k0 = kp << 1;
            int gk0 = kt + k0;
            int gn  = tileN + n;
            float v0 = (gk0 < K && gn < F) ? W[(long)gk0 * F + gn] : 0.0f;
            float v1 = (gk0 + 1 < K && gn < F) ? W[(long)(gk0 + 1) * F + gn] : 0.0f;
            unsigned int pk = (unsigned int)f32_to_bf16(v0) |
                              ((unsigned int)f32_to_bf16(v1) << 16);
            int c    = n >> 4;
            int ln   = (n & 15) + ((kp >> 3) << 4);
            int slot = k0 & 15;                     // even -> dword aligned
            int uidx = (((c << 5) + ln) << 4) + slot;
            ((unsigned int*)&sB[buf][0])[uidx >> 1] = pk;
        }
    };

    stageB(0, 0);
    int cur = 0;

    for (int kt = 0; kt < K; kt += 32) {
        __syncthreads();
        if (kt + 32 < K) stageB(cur ^ 1, kt + 32);

        // ---- A fragment direct from global (contiguous k = 16*(lane>>4) + slot) ----
        FragBF a;
        {
            int  m     = lane & 15;
            long row   = (long)tileM + wave * 16 + m;
            bool rowok = row < M;
            const float* ap = A + row * (long)K;
            int kb = kt + ((lane >> 4) << 4);       // 0 or 16 within the tile
#pragma unroll
            for (int half = 0; half < 2; ++half) {
                int ks = kb + half * 8;
                if (rowok && kvec && (ks + 8 <= K)) {
                    float4 f0 = *(const float4*)(ap + ks);
                    float4 f1 = *(const float4*)(ap + ks + 4);
                    a.u[half * 8 + 0] = f32_to_bf16(f0.x);
                    a.u[half * 8 + 1] = f32_to_bf16(f0.y);
                    a.u[half * 8 + 2] = f32_to_bf16(f0.z);
                    a.u[half * 8 + 3] = f32_to_bf16(f0.w);
                    a.u[half * 8 + 4] = f32_to_bf16(f1.x);
                    a.u[half * 8 + 5] = f32_to_bf16(f1.y);
                    a.u[half * 8 + 6] = f32_to_bf16(f1.z);
                    a.u[half * 8 + 7] = f32_to_bf16(f1.w);
                } else {
#pragma unroll
                    for (int j = 0; j < 8; ++j) {
                        int k = ks + j;
                        float v = (rowok && k < K) ? ap[k] : 0.0f;
                        a.u[half * 8 + j] = f32_to_bf16(v);
                    }
                }
            }
        }

        // ---- 4 col tiles: one wide LDS fragment load + WMMA each ----
#pragma unroll
        for (int c = 0; c < 4; ++c) {
            FragBF b;
            b.v = *reinterpret_cast<const v16bf*>(
                &sB[cur][(((c << 5) + lane) << 4)]);
            acc[c] = __builtin_amdgcn_wmma_f32_16x16x32_bf16(
                false, a.v, false, b.v, (short)0, acc[c], false, false);
        }
        cur ^= 1;
    }

    // Store D: VGPR r holds row r (lanes 0-15) / row 8+r (lanes 16-31), col = lane&15.
#pragma unroll
    for (int c = 0; c < 4; ++c) {
        int col = tileN + c * 16 + (lane & 15);
        if (col >= F) continue;
        float bv = bias ? bias[col] : 0.0f;
#pragma unroll
        for (int r = 0; r < 8; ++r) {
            int row = tileM + wave * 16 + r + 8 * (lane >> 4);
            if (row < M) C[(long)row * F + col] = acc[c][r] + bv;
        }
    }
}

// ---------------------------------------------------------------------------
// Host-side orchestration
// ---------------------------------------------------------------------------
static inline long cdiv(long a, long b) { return (a + b - 1) / b; }

extern "C" void kernel_launch(void* const* d_in, const int* in_sizes, int n_in,
                              void* d_out, int out_size, void* d_ws, size_t ws_size,
                              hipStream_t stream) {
    const long N = 100000, E = 3200000, G = 3200;

    const float* x  = (const float*)d_in[0];
    const float* W1 = (const float*)d_in[1];
    const float* b1 = (const float*)d_in[2];
    const float* W2 = (const float*)d_in[3];
    const float* b2 = (const float*)d_in[4];
    const float* W3 = (const float*)d_in[5];
    const float* b3 = (const float*)d_in[6];
    const float* Wp = (const float*)d_in[7];
    const float* bp = (const float*)d_in[8];
    const int* edge_index = (const int*)d_in[9];
    const int* src   = edge_index;
    const int* dst   = edge_index + E;
    const int* batch = (const int*)d_in[10];
    float* out = (float*)d_out;

    // Workspace carve-up (256B aligned slices)
    char* w = (char*)d_ws;
    const size_t SZ_NODE = (size_t)N * 256 * sizeof(float);   // 102,400,000 (256-divisible)
    float* bufA   = (float*)w; w += SZ_NODE;
    float* bufB   = (float*)w; w += SZ_NODE;
    float* bufC   = (float*)w; w += SZ_NODE;
    float* dis    = (float*)w; w += 400128;                    // N floats, padded
    float* nrm    = (float*)w; w += (size_t)E * sizeof(float);
    float* pooled = (float*)w; w += (size_t)G * 256 * sizeof(float);
    float* cnt    = (float*)w;

    const int T = 256;

    // --- degrees -> dis -> per-edge norm ---
    zero_f32_kernel<<<cdiv(N, T), T, 0, stream>>>(dis, N);
    deg_accum_kernel<<<cdiv(E, T), T, 0, stream>>>(dst, dis, (int)E);
    rsqrt_kernel<<<cdiv(N, T), T, 0, stream>>>(dis, (int)N);
    norm_kernel<<<cdiv(E, T), T, 0, stream>>>(src, dst, dis, nrm, (int)E);

    // --- three GCN layers ---
    const float* h_in = x;
    float* hW  = bufB;
    float* agg = bufC;
    int Kdims[3] = {37, 128, 256};
    int Fdims[3] = {128, 256, 256};
    const float* Ws[3] = {W1, W2, W3};
    const float* bs[3] = {b1, b2, b3};

    for (int L = 0; L < 3; ++L) {
        int K = Kdims[L], F = Fdims[L];
        int fshift = (F == 128) ? 7 : 8;
        int qshift = fshift - 2;

        dim3 ggrid((unsigned)cdiv(F, 64), (unsigned)cdiv(N, 128));
        gemm_wmma_bf16_kernel<<<ggrid, T, 0, stream>>>(h_in, Ws[L], nullptr, hW, (int)N, K, F);

        long totNF = N * (long)F;
        self_bias_kernel<<<cdiv(totNF, T), T, 0, stream>>>(hW, dis, bs[L], agg, totNF, fshift, F - 1);

        long totEQ = E * (long)(F / 4);
        edge_scatter_kernel<<<cdiv(totEQ, T), T, 0, stream>>>(src, dst, nrm, hW, agg,
                                                             totEQ, qshift, (F / 4) - 1, F);

        relu_kernel<<<cdiv(totNF, T), T, 0, stream>>>(agg, totNF);

        // rotate buffers: agg becomes next layer's input
        h_in = agg;
        float* next_agg = (L == 0) ? bufA : ((L == 1) ? bufC : bufB);
        hW  = (hW == bufB) ? ((next_agg == bufB) ? bufC : bufB) : bufB;
        if (hW == next_agg) hW = (next_agg == bufB) ? bufC : bufB;
        agg = next_agg;
    }
    const float* h_final = h_in;   // [N, 256], post-relu layer 3

    // --- global mean pool ---
    zero_f32_kernel<<<cdiv(G * 256, T), T, 0, stream>>>(pooled, G * 256);
    zero_f32_kernel<<<cdiv(G, T), T, 0, stream>>>(cnt, G);
    long totPQ = N * 64;
    pool_scatter_kernel<<<cdiv(totPQ, T), T, 0, stream>>>(h_final, batch, pooled, totPQ, 6, 63, 256);
    count_kernel<<<cdiv(N, T), T, 0, stream>>>(batch, cnt, (int)N);
    pool_div_kernel<<<cdiv(G * 256, T), T, 0, stream>>>(pooled, cnt, G * 256, 8);

    // --- final projection: out = pooled @ Wp + bp ---
    dim3 pgrid((unsigned)cdiv(256, 64), (unsigned)cdiv(G, 128));
    gemm_wmma_bf16_kernel<<<pgrid, T, 0, stream>>>(pooled, Wp, bp, out, (int)G, 256, 256);
}
